// Int8Linear_11029476016453
// MI455X (gfx1250) — 8-bit matrix operand order repaired and checked by reading, compiled, not run on hardware
//
#include <hip/hip_runtime.h>
#include <stdint.h>


typedef int v8i __attribute__((ext_vector_type(8)));

#define BLK 64
#define LDS_STRIDE 20

#if defined(__has_builtin)
#  if __has_builtin(__builtin_amdgcn_global_load_async_to_lds_b128) && \
      __has_builtin(__builtin_amdgcn_s_wait_asynccnt)
#    define USE_ASYNC 1
#  endif
#endif
#ifndef USE_ASYNC
#  define USE_ASYNC 0
#endif

#if USE_ASYNC
typedef int v4i_vs __attribute__((vector_size(16)));
typedef __attribute__((address_space(1))) v4i_vs* gv4p;
typedef __attribute__((address_space(3))) v4i_vs* lv4p;
#define GL_ASYNC_B128(gsrc, ldst) \
    __builtin_amdgcn_global_load_async_to_lds_b128((gv4p)(gsrc), (lv4p)(ldst), 0, 0)
#endif

__device__ __forceinline__ float bf2f(uint32_t h) {
    union { uint32_t u; float f; } c; c.u = h << 16; return c.f;
}
__device__ __forceinline__ uint16_t f2bf(float f) {
    union { float f; uint32_t u; } c; c.f = f;
    uint32_t u = c.u;
    uint32_t r = (u + 0x7fffu + ((u >> 16) & 1u)) >> 16;
    if ((u & 0x7f800000u) == 0x7f800000u) r = u >> 16;
    return (uint16_t)r;
}

__global__ __launch_bounds__(256) void quant_kernel(
    const uint16_t* __restrict__ x, int8_t* __restrict__ xq,
    float* __restrict__ xs, int K, int kb_cnt)
{
    __shared__ float red[256];
    const int t    = threadIdx.x;
    const int mblk = blockIdx.x / kb_cnt;
    const int kb   = blockIdx.x % kb_cnt;
    const int r    = t >> 2;
    const int seg  = t & 3;

    const size_t base = (size_t)(mblk * BLK + r) * (size_t)K + kb * BLK + seg * 16;
    const uint4* p = (const uint4*)(x + base);
    uint4 d0 = p[0], d1 = p[1];
    uint32_t w[8] = { d0.x, d0.y, d0.z, d0.w, d1.x, d1.y, d1.z, d1.w };

    float v[16];
    float amax = 0.0f;
#pragma unroll
    for (int i = 0; i < 8; ++i) {
        v[2 * i + 0] = bf2f(w[i] & 0xffffu);
        v[2 * i + 1] = bf2f(w[i] >> 16);
        amax = fmaxf(amax, fmaxf(fabsf(v[2 * i]), fabsf(v[2 * i + 1])));
    }

    red[t] = amax;
    __syncthreads();
#pragma unroll
    for (int s = 128; s > 0; s >>= 1) {
        if (t < s) red[t] = fmaxf(red[t], red[t + s]);
        __syncthreads();
    }
    const float am  = fmaxf(red[0], 1e-8f);
    const float inv = 127.0f / am;

    uint32_t q[4];
#pragma unroll
    for (int i = 0; i < 4; ++i) {
        uint32_t acc = 0;
#pragma unroll
        for (int j = 0; j < 4; ++j) {
            float z = v[4 * i + j] * inv;
            int qi = (int)(z + (z >= 0.0f ? 0.5f : -0.5f));
            acc |= ((uint32_t)qi & 0xffu) << (8 * j);
        }
        q[i] = acc;
    }
    *(uint4*)(xq + base) = make_uint4(q[0], q[1], q[2], q[3]);

    if (t == 0) xs[(size_t)mblk * kb_cnt + kb] = am * (1.0f / 127.0f);
}

__global__ __launch_bounds__(256, 2) void gemm_kernel(
    const int8_t* __restrict__ xq, const float* __restrict__ xs,
    const int8_t* __restrict__ wq, const float* __restrict__ ws,
    const uint16_t* __restrict__ bias, uint16_t* __restrict__ out,
    int N, int K, int kb_cnt, int nb2_cnt)
{
    __shared__ uint32_t lA[2][BLK * LDS_STRIDE];
    __shared__ uint32_t lB[2][2 * BLK * LDS_STRIDE];

    const int nblk = blockIdx.x % nb2_cnt;
    const int mblk = blockIdx.x / nb2_cnt;
    const int t    = threadIdx.x;
    const int lane = t & 31;
    const int wv   = t >> 5;
    const int mt   = wv & 3;
    const int nh   = wv >> 2;

    const int lr   = t >> 2;
    const int lseg = t & 3;
    const int8_t* gA  = xq + (size_t)(mblk * BLK + lr) * (size_t)K + lseg * 16;
    const int8_t* gB0 = wq + (size_t)(nblk * 2 * BLK + lr) * (size_t)K + lseg * 16;
    const int8_t* gB1 = gB0 + (size_t)BLK * (size_t)K;
    const int oA  = lr * LDS_STRIDE + lseg * 4;
    const int oB0 = lr * LDS_STRIDE + lseg * 4;
    const int oB1 = (lr + BLK) * LDS_STRIDE + lseg * 4;

    const int aoff = (mt * 16 + (lane & 15)) * LDS_STRIDE + ((lane >> 4) << 2);
    int boff[4];
#pragma unroll
    for (int j = 0; j < 4; ++j)
        boff[j] = (nh * 64 + j * 16 + (lane & 15)) * LDS_STRIDE + ((lane >> 4) << 2);

    float facc[4][8];
#pragma unroll
    for (int j = 0; j < 4; ++j)
#pragma unroll
        for (int i = 0; i < 8; ++i) facc[j][i] = 0.0f;

    const float* xsrow  = xs + (size_t)mblk * kb_cnt;
    const float* wsrow0 = ws + (size_t)(nblk * 2) * kb_cnt;
    const float* wsrow1 = wsrow0 + kb_cnt;

#if USE_ASYNC
    auto stage = [&](int kb, int nxt) {
        __syncthreads();
        if (kb + 1 < kb_cnt) {
            GL_ASYNC_B128(gA  + (kb + 1) * BLK, &lA[nxt][oA]);
            GL_ASYNC_B128(gB0 + (kb + 1) * BLK, &lB[nxt][oB0]);
            GL_ASYNC_B128(gB1 + (kb + 1) * BLK, &lB[nxt][oB1]);
            __builtin_amdgcn_s_wait_asynccnt(3);
        } else {
            __builtin_amdgcn_s_wait_asynccnt(0);
        }
        __syncthreads();
    };
    GL_ASYNC_B128(gA,  &lA[0][oA]);
    GL_ASYNC_B128(gB0, &lB[0][oB0]);
    GL_ASYNC_B128(gB1, &lB[0][oB1]);
#else
    auto stage = [&](int kb, int nxt) {
        uint4 av  = *(const uint4*)(gA  + kb * BLK);
        uint4 bv0 = *(const uint4*)(gB0 + kb * BLK);
        uint4 bv1 = *(const uint4*)(gB1 + kb * BLK);
        const int cur = nxt ^ 1;
        __syncthreads();
        *(uint4*)&lA[cur][oA]  = av;
        *(uint4*)&lB[cur][oB0] = bv0;
        *(uint4*)&lB[cur][oB1] = bv1;
        __syncthreads();
    };
#endif

    auto mm = [&](int cur, v8i (&dres)[4]) {
        const uint32_t* bA = lA[cur];
        const uint32_t* bB = lB[cur];
        v8i a = { (int)bA[aoff + 0], (int)bA[aoff + 1],
                  (int)bA[aoff + 2], (int)bA[aoff + 3],
                  (int)bA[aoff + 8], (int)bA[aoff + 9],
                  (int)bA[aoff + 10], (int)bA[aoff + 11] };
#pragma unroll
        for (int j = 0; j < 4; ++j) {
            const uint32_t* rb = bB + boff[j];
            v8i bb = { (int)rb[0], (int)rb[1], (int)rb[2],  (int)rb[3],
                       (int)rb[8], (int)rb[9], (int)rb[10], (int)rb[11] };
            v8i c0 = { 0, 0, 0, 0, 0, 0, 0, 0 };
            dres[j] = __builtin_amdgcn_wmma_i32_16x16x64_iu8(true, a, true, bb, c0, false, false);
        }
    };
    auto scale_at = [&](int kb) -> float {
        return xsrow[kb] * (nh ? wsrow1[kb] : wsrow0[kb]);
    };
    auto fixup = [&](const v8i (&dres)[4], float s) {
#pragma unroll
        for (int j = 0; j < 4; ++j)
#pragma unroll
            for (int i = 0; i < 8; ++i)
                facc[j][i] = fmaf((float)dres[j][i], s, facc[j][i]);
    };

    v8i dA[4], dB[4];
    float sA = 0.0f, sB = 0.0f;
    int kb = 0;
    for (; kb + 1 < kb_cnt; kb += 2) {
        stage(kb, 1);
        mm(0, dA);
        sA = scale_at(kb);
        if (kb > 0) fixup(dB, sB);
        stage(kb + 1, 0);
        mm(1, dB);
        sB = scale_at(kb + 1);
        fixup(dA, sA);
    }
    if (kb < kb_cnt) {
        stage(kb, 1);
        mm(0, dA);
        sA = scale_at(kb);
        if (kb > 0) fixup(dB, sB);
        fixup(dA, sA);
    } else if (kb_cnt > 0) {
        fixup(dB, sB);
    }

    const int colbase = nblk * 2 * BLK + nh * 64 + (lane & 15);
#pragma unroll
    for (int j = 0; j < 4; ++j) {
        const float bia = bf2f(bias[colbase + j * 16]);
#pragma unroll
        for (int i = 0; i < 8; ++i) {
            const int rowg = mblk * BLK + mt * 16 + i + ((lane >> 4) << 3);
            out[(size_t)rowg * (size_t)N + colbase + j * 16] = f2bf(facc[j][i] + bia);
        }
    }
}

extern "C" void kernel_launch(void* const* d_in, const int* in_sizes, int n_in,
                              void* d_out, int out_size, void* d_ws, size_t ws_size,
                              hipStream_t stream) {
    const uint16_t* x    = (const uint16_t*)d_in[0];
    const int8_t*   wq   = (const int8_t*)  d_in[1];
    const float*    ws   = (const float*)   d_in[2];
    const uint16_t* bias = (const uint16_t*)d_in[3];
    uint16_t*       out  = (uint16_t*)d_out;

    const int N = in_sizes[3];
    const int K = in_sizes[1] / N;
    const int M = in_sizes[0] / K;
    const int kb_cnt = K / BLK, mb_cnt = M / BLK;
    const int nb2_cnt = N / (2 * BLK);

    int8_t* xq = (int8_t*)d_ws;
    float*  xs = (float*)((char*)d_ws + (size_t)M * (size_t)K);

    quant_kernel<<<mb_cnt * kb_cnt, 256, 0, stream>>>(x, xq, xs, K, kb_cnt);
    gemm_kernel<<<mb_cnt * nb2_cnt, 256, 0, stream>>>(xq, xs, wq, ws, bias, out,
                                                      N, K, kb_cnt, nb2_cnt);
}
